// QGCN_53961969107345
// MI455X (gfx1250) — matrix operand order repaired and checked by reading, not compiled, not run on hardware
//
#include <hip/hip_runtime.h>
#include <hip/hip_bf16.h>
#include <math.h>

#define N_NODES 4096
#define F_IN    64
#define F_TOT   128
#define NHEADS  8
#define NHID    8
#define HEADDIM 64      // NHEADS*NHID
#define NCLASS  16
#define E_DIM   32
#define NSTEPS  (N_NODES / 32)   // 128 k-steps of 32

typedef __attribute__((ext_vector_type(16))) _Float16 v16h;
typedef __attribute__((ext_vector_type(8)))  float    v8f;

// ---------------- WMMA fragment helpers (CDNA5 16x16x32 f16 layouts) ----------------
// A (16xK f16): lane holds row = m0+(lane&15); halves 0..7 -> K = kb..kb+7,
// halves 8..15 -> K = kb+16..kb+23, where kb = k0 + (lane>=16 ? 8 : 0).
__device__ __forceinline__ v16h ld_a_frag_f16(const _Float16* src, int ld, int m0, int k0, int lane) {
  int row = m0 + (lane & 15);
  int kb  = k0 + ((lane & 16) ? 8 : 0);
  const _Float16* p = src + (size_t)row * ld + kb;
  v16h a;
#pragma unroll
  for (int i = 0; i < 8; ++i) a[i] = p[i];
#pragma unroll
  for (int i = 0; i < 8; ++i) a[8 + i] = p[16 + i];
  return a;
}

// B (32x16 f16): lane holds col = n0+(lane&15); halves 0..7 -> K = kb..kb+7,
// halves 8..15 -> K = kb+16..kb+23, where kb = k0 + (lane>=16 ? 8 : 0).
__device__ __forceinline__ v16h ld_b_frag_f16(const _Float16* src, int ld, int k0, int n0, int lane) {
  int col = n0 + (lane & 15);
  int kb  = k0 + ((lane & 16) ? 8 : 0);
  v16h b;
#pragma unroll
  for (int i = 0; i < 16; ++i) b[i] = src[(size_t)(kb + i + ((i >= 8) ? 8 : 0)) * ld + col];
  return b;
}

__device__ __forceinline__ v8f wmma_f16(v16h a, v16h b, v8f c) {
  return __builtin_amdgcn_wmma_f32_16x16x32_f16(false, a, false, b, (short)0, c, false, false);
}

// ---------------- feature build: x = concat(x0, emb0[e0], emb1[e1]) ----------------
__global__ void k_features(const float* __restrict__ x0, const int* __restrict__ embed,
                           const float* __restrict__ emb0, const float* __restrict__ emb1,
                           float* __restrict__ x, _Float16* __restrict__ x16) {
  int t = blockIdx.x * blockDim.x + threadIdx.x;
  if (t >= N_NODES * F_TOT) return;
  int i = t >> 7, f = t & 127;
  float v;
  if (f < F_IN)                v = x0[i * F_IN + f];
  else if (f < F_IN + E_DIM)   v = emb0[embed[i * 2 + 0] * E_DIM + (f - F_IN)];
  else                         v = emb1[embed[i * 2 + 1] * E_DIM + (f - F_IN - E_DIM)];
  x[t] = v;
  x16[t] = (_Float16)v;
}

// ---------------- weight prep: Wcat16 (128x64, head-major cols), Wout16 (64x16) ----------------
__global__ void k_prep_w(const float* __restrict__ W_heads, const float* __restrict__ W_out,
                         _Float16* __restrict__ Wcat16, _Float16* __restrict__ Wout16) {
  int t = blockIdx.x * blockDim.x + threadIdx.x;
  if (t < F_TOT * HEADDIM) {
    int k = t / HEADDIM, n = t % HEADDIM;
    int h = n >> 3, c = n & 7;
    Wcat16[t] = (_Float16)W_heads[((size_t)h * F_TOT + k) * NHID + c];   // W_heads (8,128,8)
  } else {
    int u = t - F_TOT * HEADDIM;
    if (u < HEADDIM * NCLASS) Wout16[u] = (_Float16)W_out[u];
  }
}

// ---------------- generic f16 WMMA GEMM: C(MxNcols) = A16(MxK) @ B16(KxNcols) ----------------
__global__ void k_gemm16(const _Float16* __restrict__ A, const _Float16* __restrict__ B,
                         float* __restrict__ C, _Float16* __restrict__ C16,
                         int M, int K, int Ncols) {
  int lane   = threadIdx.x & 31;
  int ntiles = Ncols >> 4;
  int tile   = blockIdx.x * (blockDim.x >> 5) + (threadIdx.x >> 5);
  if (tile >= (M >> 4) * ntiles) return;
  int m0 = (tile / ntiles) << 4;
  int n0 = (tile % ntiles) << 4;
  v8f acc = {};
  for (int k0 = 0; k0 < K; k0 += 32) {
    v16h a = ld_a_frag_f16(A, K, m0, k0, lane);
    v16h b = ld_b_frag_f16(B, Ncols, k0, n0, lane);
    acc = wmma_f16(a, b, acc);
  }
  int col   = n0 + (lane & 15);
  int rbase = m0 + ((lane & 16) ? 8 : 0);
#pragma unroll
  for (int v = 0; v < 8; ++v) {
    float r = acc[v];
    C[(size_t)(rbase + v) * Ncols + col] = r;
    C16[(size_t)(rbase + v) * Ncols + col] = (_Float16)r;
  }
}

// ---------------- per-head attention scalars: f = Wh.a[:8], g = Wh.a[8:] ----------------
__global__ void k_fg_heads(const float* __restrict__ Wh, const float* __restrict__ a_heads,
                           float* __restrict__ f_all, float* __restrict__ g_all) {
  int t = blockIdx.x * blockDim.x + threadIdx.x;
  if (t >= NHEADS * N_NODES) return;
  int h = t / N_NODES, i = t % N_NODES;
  const float* w = Wh + (size_t)i * HEADDIM + h * NHID;
  const float* a = a_heads + h * 2 * NHID;              // a_heads (8,16,1)
  float f = 0.f, g = 0.f;
#pragma unroll
  for (int c = 0; c < NHID; ++c) { f += w[c] * a[c]; g += w[c] * a[NHID + c]; }
  f_all[t] = f; g_all[t] = g;
}

__global__ void k_fg_out(const float* __restrict__ Wh2, const float* __restrict__ a_out,
                         float* __restrict__ f2, float* __restrict__ g2) {
  int i = blockIdx.x * blockDim.x + threadIdx.x;
  if (i >= N_NODES) return;
  const float* w = Wh2 + (size_t)i * NCLASS;
  float f = 0.f, g = 0.f;
#pragma unroll
  for (int c = 0; c < NCLASS; ++c) { f += w[c] * a_out[c]; g += w[c] * a_out[NCLASS + c]; }
  f2[i] = f; g2[i] = g;
}

// ---------------- packed B fragments (per-lane WMMA-B register order) ----------------
// Heads: B tile per (head, kstep): cols 0..7 = Wh head cols, col 8 = 1.0 (Z), rest 0.
// Layout: Bpack[((h*NSTEPS + ks)*32 + lane)*16 + i]  ->  each lane loads one aligned v16h.
__global__ void k_bpack_heads(const _Float16* __restrict__ Wh16, _Float16* __restrict__ Bpack) {
  int t = blockIdx.x * blockDim.x + threadIdx.x;        // NHEADS*NSTEPS*32 threads
  if (t >= NHEADS * NSTEPS * 32) return;
  int lane = t & 31, ks = (t >> 5) & (NSTEPS - 1), h = t >> 12;
  int col = lane & 15;
  int kbb = ks * 32 + ((lane & 16) ? 8 : 0);
  _Float16* out = Bpack + (size_t)t * 16;
#pragma unroll
  for (int i = 0; i < 16; ++i) {
    _Float16 v = (_Float16)0.0f;
    if (col < NHID)       v = Wh16[(size_t)(kbb + i + ((i >= 8) ? 8 : 0)) * HEADDIM + h * NHID + col];
    else if (col == NHID) v = (_Float16)1.0f;
    out[i] = v;
  }
}

// Out layer: B tile per kstep: cols 0..15 = Wh2 columns.
__global__ void k_bpack_out(const _Float16* __restrict__ Wh216, _Float16* __restrict__ Bpack) {
  int t = blockIdx.x * blockDim.x + threadIdx.x;        // NSTEPS*32 threads
  if (t >= NSTEPS * 32) return;
  int lane = t & 31, ks = t >> 5;
  int col = lane & 15;
  int kbb = ks * 32 + ((lane & 16) ? 8 : 0);
  _Float16* out = Bpack + (size_t)t * 16;
#pragma unroll
  for (int i = 0; i < 16; ++i)
    out[i] = Wh216[(size_t)(kbb + i + ((i >= 8) ? 8 : 0)) * NCLASS + col];
}

// ---------------- fused 8-head attention: one pass over A per 16-row tile ----------------
// P(i,j) = A(i,j) * exp(tanh(s) - 2/(s^2+1)),  s = f_i + g_j  (bounded -> no max shift).
__global__ void __launch_bounds__(32) k_attn_heads(const float* __restrict__ Aadj,
                                                   const _Float16* __restrict__ Bpack,
                                                   const float* __restrict__ f_all,
                                                   const float* __restrict__ g_all,
                                                   _Float16* __restrict__ xh16) {
  int lane = threadIdx.x & 31;
  int m0   = blockIdx.x << 4;
  int arow = m0 + (lane & 15);
  float fh[NHEADS];
#pragma unroll
  for (int h = 0; h < NHEADS; ++h) fh[h] = f_all[h * N_NODES + arow];

  v8f zero = {};
  v8f acc[NHEADS];
#pragma unroll
  for (int h = 0; h < NHEADS; ++h) acc[h] = zero;

  const v16h* bvec = (const v16h*)Bpack;

  for (int k0 = 0; k0 < N_NODES; k0 += 32) {
    int ks = k0 >> 5;
    int kb = k0 + ((lane & 16) ? 8 : 0);
    const float* ap = Aadj + (size_t)arow * N_NODES + kb;
    __builtin_prefetch(ap + 32, 0, 1);                  // global_prefetch hint
    float4 q0 = *(const float4*)(ap);
    float4 q1 = *(const float4*)(ap + 4);
    float4 q2 = *(const float4*)(ap + 16);
    float4 q3 = *(const float4*)(ap + 20);
    float av[16] = { q0.x, q0.y, q0.z, q0.w, q1.x, q1.y, q1.z, q1.w,
                     q2.x, q2.y, q2.z, q2.w, q3.x, q3.y, q3.z, q3.w };
#pragma unroll
    for (int h = 0; h < NHEADS; ++h) {
      v16h p;
#pragma unroll
      for (int i = 0; i < 16; ++i) {
        float a = av[i];
        float pv = 0.f;
        if (a != 0.f) {                                  // ~1% density: skip transcendentals
          int j = kb + i + ((i >= 8) ? 8 : 0);           // kb+i (i<8), kb+16+(i-8)
          float s = fh[h] + g_all[h * N_NODES + j];
          float e = tanhf(s) - 2.f / (s * s + 1.f);
          pv = a * __expf(e);                            // A in {0,1}: a*exp(e) == exp(e)
        }
        p[i] = (_Float16)pv;
      }
      v16h b = bvec[(size_t)(h * NSTEPS + ks) * 32 + lane];   // 2x global_load_b128
      acc[h] = wmma_f16(p, b, acc[h]);
    }
  }

  // epilogue: hp = acc/Z, head activation elu(hp) - 2/(hp^2+1)
  int col   = lane & 15;
  int rbase = m0 + ((lane & 16) ? 8 : 0);
  int zsrc  = (lane & 16) ? (16 + NHID) : NHID;          // lane holding the ones-column
#pragma unroll
  for (int h = 0; h < NHEADS; ++h) {
#pragma unroll
    for (int v = 0; v < 8; ++v) {
      float val = acc[h][v];
      float z   = __shfl(val, zsrc, 32);
      if (col < NHID) {
        float hp = val / z;
        float e  = (hp > 0.f) ? hp : (__expf(hp) - 1.f);
        float r  = e - 2.f / (hp * hp + 1.f);
        xh16[(size_t)(rbase + v) * HEADDIM + h * NHID + col] = (_Float16)r;
      }
    }
  }
}

// ---------------- output attention layer: x1 = elu((P2 @ Wh2) / Z) ----------------
__global__ void __launch_bounds__(32) k_attn_out(const float* __restrict__ Aadj,
                                                 const _Float16* __restrict__ Bpack,
                                                 const float* __restrict__ f2,
                                                 const float* __restrict__ g2,
                                                 float* __restrict__ x1) {
  int lane = threadIdx.x & 31;
  int m0   = blockIdx.x << 4;
  int arow = m0 + (lane & 15);
  float fr = f2[arow];
  v8f acc = {}, accz = {};
  int col = lane & 15;
  const v16h* bvec = (const v16h*)Bpack;

  v16h bz;
#pragma unroll
  for (int i = 0; i < 16; ++i) bz[i] = (col == 0) ? (_Float16)1.0f : (_Float16)0.0f;

  for (int k0 = 0; k0 < N_NODES; k0 += 32) {
    int ks = k0 >> 5;
    int kb = k0 + ((lane & 16) ? 8 : 0);
    const float* ap = Aadj + (size_t)arow * N_NODES + kb;
    __builtin_prefetch(ap + 32, 0, 1);
    float4 q0 = *(const float4*)(ap);
    float4 q1 = *(const float4*)(ap + 4);
    float4 q2 = *(const float4*)(ap + 16);
    float4 q3 = *(const float4*)(ap + 20);
    float av[16] = { q0.x, q0.y, q0.z, q0.w, q1.x, q1.y, q1.z, q1.w,
                     q2.x, q2.y, q2.z, q2.w, q3.x, q3.y, q3.z, q3.w };
    v16h p;
#pragma unroll
    for (int i = 0; i < 16; ++i) {
      float a = av[i];
      float pv = 0.f;
      if (a != 0.f) {
        int j = kb + i + ((i >= 8) ? 8 : 0);
        float s = fr + g2[j];
        float e = tanhf(s) - 2.f / (s * s + 1.f);
        pv = a * __expf(e);
      }
      p[i] = (_Float16)pv;
    }
    v16h b = bvec[(size_t)ks * 32 + lane];
    acc  = wmma_f16(p, b,  acc);
    accz = wmma_f16(p, bz, accz);
  }
  int rbase = m0 + ((lane & 16) ? 8 : 0);
  int zsrc  = (lane & 16) ? 16 : 0;
#pragma unroll
  for (int v = 0; v < 8; ++v) {
    float z  = __shfl(accz[v], zsrc, 32);
    float hp = acc[v] / z;
    float r  = (hp > 0.f) ? hp : (__expf(hp) - 1.f);     // elu only (concat=False)
    x1[(size_t)(rbase + v) * NCLASS + col] = r;
  }
}

// ---------------- final collapse ----------------
__global__ void k_u(const float* __restrict__ x1, const float* __restrict__ left_w,
                    float* __restrict__ u) {
  int i = blockIdx.x * blockDim.x + threadIdx.x;
  if (i >= N_NODES) return;
  float s = 0.f;
#pragma unroll
  for (int c = 0; c < NCLASS; ++c) s += x1[(size_t)i * NCLASS + c] * left_w[c];
  u[i] = s;
}

__global__ void k_lhs(const float* __restrict__ Aadj, const float* __restrict__ u,
                      const float* __restrict__ left_b, float* __restrict__ lhs) {
  int i = blockIdx.x * blockDim.x + threadIdx.x;
  if (i >= N_NODES) return;
  float s = 0.f;
  for (int k = 0; k < N_NODES; ++k) s += Aadj[(size_t)k * N_NODES + i] * u[k];   // coalesced over i
  lhs[i] = s + left_b[0];
}

__global__ void k_final(const float* __restrict__ lhs, const float* __restrict__ x,
                        const float* __restrict__ right_w, const float* __restrict__ right_b,
                        float* __restrict__ out) {
  __shared__ float sm[F_TOT];
  int f = threadIdx.x;                    // 128 threads
  float v = 0.f;
  for (int i = 0; i < N_NODES; ++i) v += lhs[i] * x[(size_t)i * F_TOT + f];
  sm[f] = v * right_w[f];
  __syncthreads();
  if (f == 0) {
    float s = 0.f;
    for (int j = 0; j < F_TOT; ++j) s += sm[j];
    out[0] = s + right_b[0];
  }
}

// ---------------- host launch ----------------
extern "C" void kernel_launch(void* const* d_in, const int* in_sizes, int n_in,
                              void* d_out, int out_size, void* d_ws, size_t ws_size,
                              hipStream_t stream) {
  const float* A       = (const float*)d_in[0];
  const float* x0      = (const float*)d_in[1];
  const int*   embed   = (const int*)  d_in[2];
  const float* emb0    = (const float*)d_in[3];
  const float* emb1    = (const float*)d_in[4];
  const float* W_heads = (const float*)d_in[5];
  const float* a_heads = (const float*)d_in[6];
  const float* W_out   = (const float*)d_in[7];
  const float* a_out   = (const float*)d_in[8];
  const float* left_w  = (const float*)d_in[9];
  const float* left_b  = (const float*)d_in[10];
  const float* right_w = (const float*)d_in[11];
  const float* right_b = (const float*)d_in[12];

  char* ws = (char*)d_ws;
  size_t off = 0;
  auto alloc = [&](size_t bytes) -> void* {
    void* p = ws + off; off += (bytes + 255) & ~(size_t)255; return p;
  };
  float*    x       = (float*)   alloc((size_t)N_NODES * F_TOT * 4);
  _Float16* x16     = (_Float16*)alloc((size_t)N_NODES * F_TOT * 2);
  _Float16* Wcat16  = (_Float16*)alloc((size_t)F_TOT * HEADDIM * 2);
  _Float16* Wout16  = (_Float16*)alloc((size_t)HEADDIM * NCLASS * 2);
  float*    Wh      = (float*)   alloc((size_t)N_NODES * HEADDIM * 4);
  _Float16* Wh16    = (_Float16*)alloc((size_t)N_NODES * HEADDIM * 2);
  float*    f_all   = (float*)   alloc((size_t)NHEADS * N_NODES * 4);
  float*    g_all   = (float*)   alloc((size_t)NHEADS * N_NODES * 4);
  _Float16* Bpack_h = (_Float16*)alloc((size_t)NHEADS * NSTEPS * 32 * 16 * 2);  // 1 MB
  _Float16* xh16    = (_Float16*)alloc((size_t)N_NODES * HEADDIM * 2);
  float*    Wh2     = (float*)   alloc((size_t)N_NODES * NCLASS * 4);
  _Float16* Wh216   = (_Float16*)alloc((size_t)N_NODES * NCLASS * 2);
  _Float16* Bpack_o = (_Float16*)alloc((size_t)NSTEPS * 32 * 16 * 2);           // 128 KB
  float*    f2      = (float*)   alloc((size_t)N_NODES * 4);
  float*    g2      = (float*)   alloc((size_t)N_NODES * 4);
  float*    x1      = (float*)   alloc((size_t)N_NODES * NCLASS * 4);
  float*    u       = (float*)   alloc((size_t)N_NODES * 4);
  float*    lhs     = (float*)   alloc((size_t)N_NODES * 4);
  (void)ws_size; (void)in_sizes; (void)n_in; (void)out_size;

  k_features<<<(N_NODES * F_TOT + 255) / 256, 256, 0, stream>>>(x0, embed, emb0, emb1, x, x16);
  k_prep_w<<<(F_TOT * HEADDIM + HEADDIM * NCLASS + 255) / 256, 256, 0, stream>>>(W_heads, W_out, Wcat16, Wout16);

  // Wh_all = x @ Wcat : 4096x128 @ 128x64 -> 1024 tiles, 8 waves/block
  k_gemm16<<<128, 256, 0, stream>>>(x16, Wcat16, Wh, Wh16, N_NODES, F_TOT, HEADDIM);
  k_fg_heads<<<(NHEADS * N_NODES + 255) / 256, 256, 0, stream>>>(Wh, a_heads, f_all, g_all);
  k_bpack_heads<<<(NHEADS * NSTEPS * 32 + 255) / 256, 256, 0, stream>>>(Wh16, Bpack_h);

  // fused 8-head attention, one 16-row tile per wave
  k_attn_heads<<<N_NODES / 16, 32, 0, stream>>>(A, Bpack_h, f_all, g_all, xh16);

  // Wh2 = xh @ W_out : 4096x64 @ 64x16 -> 256 tiles
  k_gemm16<<<32, 256, 0, stream>>>(xh16, Wout16, Wh2, Wh216, N_NODES, HEADDIM, NCLASS);
  k_fg_out<<<(N_NODES + 255) / 256, 256, 0, stream>>>(Wh2, a_out, f2, g2);
  k_bpack_out<<<(NSTEPS * 32 + 255) / 256, 256, 0, stream>>>(Wh216, Bpack_o);
  k_attn_out<<<N_NODES / 16, 32, 0, stream>>>(A, Bpack_o, f2, g2, x1);

  k_u<<<N_NODES / 256, 256, 0, stream>>>(x1, left_w, u);
  k_lhs<<<N_NODES / 256, 256, 0, stream>>>(A, u, left_b, lhs);
  k_final<<<1, F_TOT, 0, stream>>>(lhs, x, right_w, right_b, (float*)d_out);
}
